// SkipCCell_61890478735976
// MI455X (gfx1250) — compile-verified, matrix operand order checked by reading, not run on hardware
//
#include <hip/hip_runtime.h>
#include <hip/hip_bf16.h>
#include <math.h>

// ---------------- types ----------------
typedef __bf16 bf16_t;
typedef __attribute__((ext_vector_type(16))) __bf16 v16bf;
typedef __attribute__((ext_vector_type(8)))  __bf16 v8bf;
typedef __attribute__((ext_vector_type(8)))  float  v8f;
typedef int v4i_t __attribute__((vector_size(16)));

// ---------------- problem constants ----------------
#define T_STEPS 512
#define BATCH   64
#define IN_DIM  1024
#define HID     1024
#define NGATES  4096          // 4*HID
#define KCAT    2048          // IN_DIM + HID
#define SKIPN   4
#define RGRID   64            // persistent blocks in recurrent kernel

// ---------------- workspace layout (bytes) ----------------
constexpr size_t OFF_ZX   = 0;                                               // fp32 (T*B, 4096)
constexpr size_t SZ_ZX    = (size_t)T_STEPS * BATCH * NGATES * 4;
constexpr size_t OFF_XB   = OFF_ZX + SZ_ZX;                                  // bf16 x
constexpr size_t SZ_XB    = (size_t)T_STEPS * BATCH * IN_DIM * 2;
constexpr size_t OFF_WB   = OFF_XB + SZ_XB;                                  // bf16 W = [w_ih | w_hh]
constexpr size_t SZ_WB    = (size_t)NGATES * KCAT * 2;
constexpr size_t OFF_BIAS = OFF_WB + SZ_WB;                                  // fp32 4096
constexpr size_t OFF_HB   = OFF_BIAS + NGATES * 4;                           // bf16 h state (B,H)
constexpr size_t OFF_CNT  = OFF_HB + (size_t)BATCH * HID * 2;                // unsigned barrier counter

// ---------------- CDNA5 async-copy path (guarded) ----------------
#if defined(__gfx1250__) && __has_builtin(__builtin_amdgcn_global_load_async_to_lds_b128)
#define HAVE_ASYNC_LDS 1
#endif

__device__ __forceinline__ void cp16_async(void* lds, const void* g) {
#ifdef HAVE_ASYNC_LDS
  __builtin_amdgcn_global_load_async_to_lds_b128(
      (__attribute__((address_space(1))) v4i_t*)(g),
      (__attribute__((address_space(3))) v4i_t*)(lds), 0, 0);
#else
  *(uint4*)lds = *(const uint4*)g;
#endif
}

#ifdef HAVE_ASYNC_LDS
#if __has_builtin(__builtin_amdgcn_s_wait_asynccnt)
#define WAIT_ASYNC(n) __builtin_amdgcn_s_wait_asynccnt(n)
#else
#define WAIT_ASYNC(n) asm volatile("s_wait_asynccnt " #n ::: "memory")
#endif
#else
#define WAIT_ASYNC(n) ((void)0)
#endif

// ---------------- WMMA helpers ----------------
// A-fragment (16x32 bf16, M x K) per ISA 7.12.2: lane<16 -> row M=lane, elems
// {k0..k0+7, k0+16..k0+23}; lane>=16 -> same rows, {k0+8.., k0+24..}.
// The SAME loader works for the B-fragment when 'base' is W row-major (N rows
// of K): lane L holds column N = L%16 of B = W^T.
__device__ __forceinline__ v16bf load_frag(const bf16_t* base, int ld, int row0, int k0) {
  const int lane = threadIdx.x & 31;
  const int half = lane >> 4;
  const int r    = lane & 15;
  const bf16_t* p = base + (size_t)(row0 + r) * ld + k0 + half * 8;
  union { v16bf v; v8bf h[2]; } u;
  u.h[0] = *(const v8bf*)(p);
  u.h[1] = *(const v8bf*)(p + 16);
  return u.v;
}

__device__ __forceinline__ v8f wmma_bf16(v16bf a, v16bf b, v8f c) {
  return __builtin_amdgcn_wmma_f32_16x16x32_bf16(false, a, false, b, (short)0, c, false, false);
}

__device__ __forceinline__ float sigf(float x) { return 1.0f / (1.0f + __expf(-x)); }

// ---------------- grid-wide barrier (persistent kernel) ----------------
__device__ __forceinline__ void grid_barrier(unsigned* counter, unsigned target) {
  __syncthreads();
  if (threadIdx.x == 0) {
    __threadfence();
    atomicAdd(counter, 1u);
    while (__hip_atomic_load(counter, __ATOMIC_RELAXED, __HIP_MEMORY_SCOPE_AGENT) < target)
      __builtin_amdgcn_s_sleep(1);
    __threadfence();
  }
  __syncthreads();
}

// ---------------- kernel 1: convert / pack ----------------
__global__ void prep_kernel(const float* __restrict__ x, const float* __restrict__ h0,
                            const float* __restrict__ w_ih, const float* __restrict__ w_hh,
                            const float* __restrict__ b_ih, const float* __restrict__ b_hh,
                            const int* __restrict__ bias_flag,
                            bf16_t* __restrict__ xb, bf16_t* __restrict__ wb,
                            float* __restrict__ biasc, bf16_t* __restrict__ hb,
                            unsigned* __restrict__ counter) {
  const size_t tid    = (size_t)blockIdx.x * blockDim.x + threadIdx.x;
  const size_t stride = (size_t)gridDim.x * blockDim.x;
  for (size_t i = tid; i < (size_t)T_STEPS * BATCH * IN_DIM; i += stride)
    xb[i] = (bf16_t)x[i];
  for (size_t i = tid; i < (size_t)NGATES * KCAT; i += stride) {
    const size_t n = i / KCAT, k = i % KCAT;
    const float v = (k < IN_DIM) ? w_ih[n * IN_DIM + k] : w_hh[n * HID + (k - IN_DIM)];
    wb[i] = (bf16_t)v;
  }
  const int bf = *bias_flag;
  for (size_t i = tid; i < (size_t)NGATES; i += stride)
    biasc[i] = bf ? (b_ih[i] + b_hh[i]) : 0.0f;
  for (size_t i = tid; i < (size_t)BATCH * HID; i += stride)
    hb[i] = (bf16_t)h0[i];
  if (tid == 0) *counter = 0u;
}

// ---------------- kernel 2: z_x = x @ w_ih^T + b ----------------
// 128x128 block tile, 16 waves, each wave a 32x32 register tile (2x2 WMMA
// tiles, 4 accumulators): 4 WMMAs per barrier pair, 4 WMMAs per 8 ds_loads.
// A/B k-tiles (128x32 bf16) double-buffered in LDS via async copies.
__global__ __launch_bounds__(512) void gemm_x_kernel(const bf16_t* __restrict__ xb,
                                                     const bf16_t* __restrict__ wb,
                                                     const float* __restrict__ biasc,
                                                     float* __restrict__ zx) {
  __shared__ bf16_t As[2][128][32];
  __shared__ bf16_t Bs[2][128][32];

  const int tid  = threadIdx.x;
  const int wave = tid >> 5;
  const int mw   = wave >> 2;         // 0..3 -> rows mw*32
  const int nw   = wave & 3;          // 0..3 -> cols nw*32
  const int mblk = blockIdx.y * 128;
  const int nblk = blockIdx.x * 128;

  auto stage = [&](int buf, int k0) {
    // 128x32 bf16 tile = 512 x 16B chunks; every thread copies one A and one B chunk
    const int row = tid >> 2;
    const int q   = tid & 3;
    cp16_async(&As[buf][row][q * 8], xb + (size_t)(mblk + row) * IN_DIM + k0 + q * 8);
    cp16_async(&Bs[buf][row][q * 8], wb + (size_t)(nblk + row) * KCAT + k0 + q * 8);
  };

  v8f acc00 = {}, acc01 = {}, acc10 = {}, acc11 = {};
  stage(0, 0);
  for (int kk = 0; kk < IN_DIM / 32; ++kk) {
    const int cur = kk & 1;
    if (kk + 1 < IN_DIM / 32) {
      stage(cur ^ 1, (kk + 1) * 32);
      WAIT_ASYNC(2);             // previous k-tile landed (2 async ops in flight)
    } else {
      WAIT_ASYNC(0);
    }
    __syncthreads();
    v16bf a0 = load_frag(&As[cur][0][0], 32, mw * 32, 0);
    v16bf a1 = load_frag(&As[cur][0][0], 32, mw * 32 + 16, 0);
    v16bf b0 = load_frag(&Bs[cur][0][0], 32, nw * 32, 0);
    v16bf b1 = load_frag(&Bs[cur][0][0], 32, nw * 32 + 16, 0);
    acc00 = wmma_bf16(a0, b0, acc00);
    acc01 = wmma_bf16(a0, b1, acc01);
    acc10 = wmma_bf16(a1, b0, acc10);
    acc11 = wmma_bf16(a1, b1, acc11);
    __syncthreads();
  }

  // C/D layout: lane<16 -> N=lane, M=m0+i; lane>=16 -> N=lane-16, M=m0+8+i
  const int lane = tid & 31;
  const int jn   = lane & 15;
  const int mtop = (lane >> 4) * 8;
  const v8f* accs[2][2] = {{&acc00, &acc01}, {&acc10, &acc11}};
#pragma unroll
  for (int im = 0; im < 2; ++im) {
#pragma unroll
    for (int in = 0; in < 2; ++in) {
      const int n  = nblk + nw * 32 + in * 16 + jn;
      const int m0 = mblk + mw * 32 + im * 16 + mtop;
      const float bs = biasc[n];
      float* zp = zx + (size_t)m0 * NGATES + n;
      const v8f& a = *accs[im][in];
#pragma unroll
      for (int i = 0; i < 8; ++i)
        zp[(size_t)i * NGATES] = a[i] + bs;
    }
  }
}

// ---------------- kernel 3: persistent recurrent LSTM ----------------
// Block j owns h-units [16j, 16j+16) => its 64 gate columns are {g*1024+16j..}
// for g in {i,f,g,o}: gates, cell state and SKIP queue stay in LDS. One atomic
// grid barrier per step publishes the new bf16 h.
__global__ __launch_bounds__(512) void recurrent_kernel(const float* __restrict__ zx,
                                                        const bf16_t* __restrict__ wb,
                                                        const float* __restrict__ c0,
                                                        bf16_t* __restrict__ hb,
                                                        float* __restrict__ out,
                                                        unsigned* __restrict__ counter) {
  __shared__ float zs[16][16][16];            // [mt*4+gate][m][j]
  __shared__ float cbuf[BATCH][16];           // cell state for this block's columns
  __shared__ float skipq[SKIPN][BATCH][16];   // skip-hidden queue

  const int tid   = threadIdx.x;
  const int wave  = tid >> 5;
  const int lane  = tid & 31;
  const int mt    = wave >> 2;                // batch row-tile (0..3)
  const int gate  = wave & 3;                 // 0=i 1=f 2=g 3=o
  const int hbase = blockIdx.x * 16;

  for (int p = tid; p < BATCH * 16; p += 512) {
    const int b = p >> 4, j = p & 15;
    cbuf[b][j] = c0[(size_t)b * HID + hbase + j];
#pragma unroll
    for (int s = 0; s < SKIPN; ++s) skipq[s][b][j] = 0.0f;
  }
  __syncthreads();

  const int jn   = lane & 15;
  const int mtop = (lane >> 4) * 8;
  // B-matrix rows: n = gate*1024 + hbase + (lane&15); K runs over w_hh cols
  const bf16_t* wrow = wb + (size_t)(gate * HID + hbase) * KCAT + IN_DIM;

  for (int t = 0; t < T_STEPS; ++t) {
    v8f acc = {};
#pragma unroll 4
    for (int k0 = 0; k0 < HID; k0 += 32) {
      v16bf a = load_frag(hb, HID, mt * 16, k0);     // h (bf16, 64x1024)
      v16bf b = load_frag(wrow, KCAT, 0, k0);        // w_hh rows (L2-resident)
      acc = wmma_bf16(a, b, acc);
    }
#pragma unroll
    for (int i = 0; i < 8; ++i)
      zs[mt * 4 + gate][mtop + i][jn] = acc[i];
    __syncthreads();

    // elementwise gates + skip queue (2 elements per thread)
    for (int p = tid; p < BATCH * 16; p += 512) {
      const int b = p >> 4, j = p & 15;
      const int bt = b >> 4, bm = b & 15;
      const size_t zbase = ((size_t)t * BATCH + b) * NGATES + hbase + j;
      const float zi = zs[bt * 4 + 0][bm][j] + zx[zbase + 0 * HID];
      const float zf = zs[bt * 4 + 1][bm][j] + zx[zbase + 1 * HID];
      const float zg = zs[bt * 4 + 2][bm][j] + zx[zbase + 2 * HID];
      const float zo = zs[bt * 4 + 3][bm][j] + zx[zbase + 3 * HID];
      float c = sigf(zf) * cbuf[b][j] + sigf(zi) * tanhf(zg);
      float h = sigf(zo) * tanhf(c);
      const int idx = t & (SKIPN - 1);
      if (t >= SKIPN) h += skipq[idx][b][j];
      skipq[idx][b][j] = h;
      cbuf[b][j] = c;
      out[(size_t)t * BATCH * HID + (size_t)b * HID + hbase + j] = h;
      hb[(size_t)b * HID + hbase + j] = (bf16_t)h;
    }
    grid_barrier(counter, (unsigned)(t + 1) * RGRID);  // all blocks see new h
  }

  // final h, c appended after the (T,B,H) sequence
  const size_t base = (size_t)T_STEPS * BATCH * HID;
  for (int p = tid; p < BATCH * 16; p += 512) {
    const int b = p >> 4, j = p & 15;
    out[base + (size_t)b * HID + hbase + j] = skipq[(T_STEPS - 1) & (SKIPN - 1)][b][j];
    out[base + (size_t)BATCH * HID + (size_t)b * HID + hbase + j] = cbuf[b][j];
  }
}

// ---------------- launch ----------------
extern "C" void kernel_launch(void* const* d_in, const int* in_sizes, int n_in,
                              void* d_out, int out_size, void* d_ws, size_t ws_size,
                              hipStream_t stream) {
  const float* x    = (const float*)d_in[0];
  const float* h0   = (const float*)d_in[1];
  const float* c0   = (const float*)d_in[2];
  const float* w_ih = (const float*)d_in[3];
  const float* w_hh = (const float*)d_in[4];
  const float* b_ih = (const float*)d_in[5];
  const float* b_hh = (const float*)d_in[6];
  const int*   bias = (const int*)d_in[7];

  char* ws = (char*)d_ws;
  float*    zx    = (float*)(ws + OFF_ZX);
  bf16_t*   xb    = (bf16_t*)(ws + OFF_XB);
  bf16_t*   wb    = (bf16_t*)(ws + OFF_WB);
  float*    biasc = (float*)(ws + OFF_BIAS);
  bf16_t*   hb    = (bf16_t*)(ws + OFF_HB);
  unsigned* cnt   = (unsigned*)(ws + OFF_CNT);

  prep_kernel<<<4096, 256, 0, stream>>>(x, h0, w_ih, w_hh, b_ih, b_hh, bias,
                                        xb, wb, biasc, hb, cnt);
  gemm_x_kernel<<<dim3(NGATES / 128, (T_STEPS * BATCH) / 128), 512, 0, stream>>>(xb, wb, biasc, zx);
  recurrent_kernel<<<RGRID, 512, 0, stream>>>(zx, wb, c0, hb, (float*)d_out, cnt);
}
